// GCNLayer_3556232921092
// MI455X (gfx1250) — matrix operand order repaired and checked by reading, not compiled, not run on hardware
//
#include <hip/hip_runtime.h>

// GCN layer on MI455X (gfx1250): out = relu( (D^-1/2 (A+I) D^-1/2 @ x) @ W )
// N=8192, F=128.  adj (256MB f32) is the only HBM-heavy operand (2 streaming
// passes => ~22us floor @ 23.3TB/s); y/W/s (~4MB) stay L2-resident.  The big
// GEMM runs as v_wmma_f32_16x16x32_bf16 with f32 accumulation, double-buffered
// LDS tiles, async global->LDS copies for the B operand, and NT hints on the
// adj stream.

#define NROWS 8192
#define NF    128

typedef __attribute__((ext_vector_type(16))) __bf16 v16bf;
typedef __attribute__((ext_vector_type(8)))  float  v8f;
typedef __attribute__((ext_vector_type(2)))  __bf16 v2bf;
typedef __attribute__((ext_vector_type(4)))  float  v4f;
typedef __attribute__((ext_vector_type(4)))  int    v4i;

typedef __attribute__((address_space(1))) v4i* gv4i_p;   // global
typedef __attribute__((address_space(3))) v4i* lv4i_p;   // LDS

// ---------------------------------------------------------------------------
// bf16 helpers (hardware packed convert when the target enables it)
// ---------------------------------------------------------------------------
__device__ __forceinline__ unsigned short f32_to_bf16(float f) {
    union { float f; unsigned u; } v; v.f = f;
    return (unsigned short)((v.u + 0x8000u) >> 16);   // round-to-nearest
}
__device__ __forceinline__ float bf16_to_f32(unsigned short h) {
    union { unsigned u; float f; } v; v.u = ((unsigned)h) << 16;
    return v.f;
}
__device__ __forceinline__ unsigned pack_bf16x2(float lo, float hi) {
#if __has_builtin(__builtin_amdgcn_cvt_pk_bf16_f32)
    union { v2bf v; unsigned u; } c;
    c.v = __builtin_amdgcn_cvt_pk_bf16_f32(lo, hi);
    return c.u;
#else
    union { float f; unsigned u; } a, b; a.f = lo; b.f = hi;
    return ((a.u + 0x8000u) >> 16) | ((b.u + 0x8000u) & 0xFFFF0000u);
#endif
}

// ---------------------------------------------------------------------------
// async global->LDS copy (ASYNCcnt path); falls back to reg-staged copy
// ---------------------------------------------------------------------------
#if __has_builtin(__builtin_amdgcn_global_load_async_to_lds_b128)
#define HAVE_ASYNC_LDS 1
#endif

__device__ __forceinline__ void fill_y32B(unsigned short* dst, const unsigned short* src) {
#if defined(HAVE_ASYNC_LDS)
    __builtin_amdgcn_global_load_async_to_lds_b128((gv4i_p)src, (lv4i_p)dst, 0, 0);
    __builtin_amdgcn_global_load_async_to_lds_b128((gv4i_p)(src + 8), (lv4i_p)(dst + 8), 0, 0);
#else
    const uint4* sp = (const uint4*)src;
    uint4 a = sp[0], b = sp[1];
    ((uint4*)dst)[0] = a;
    ((uint4*)dst)[1] = b;
#endif
}

__device__ __forceinline__ void wait_async_and_barrier() {
#if defined(HAVE_ASYNC_LDS)
#if __has_builtin(__builtin_amdgcn_s_wait_asynccnt)
    __builtin_amdgcn_s_wait_asynccnt(0);
#else
    asm volatile("s_wait_asynccnt 0" ::: "memory");
#endif
#endif
    __syncthreads();
}

// ---------------------------------------------------------------------------
// 1) dis[i] = rsqrt( rowsum(adj[i,:]) + 1 + 1e-8 )
// ---------------------------------------------------------------------------
__global__ void k_deg(const float* __restrict__ adj, float* __restrict__ dis) {
    __shared__ float red[256];
    const int row = blockIdx.x;
    const int t   = threadIdx.x;
    const v4f* av = (const v4f*)(adj + (size_t)row * NROWS);
    float s = 0.f;
#pragma unroll
    for (int j = 0; j < 8; ++j) {
        v4f f = __builtin_nontemporal_load(&av[t + j * 256]);
        s += f.x + f.y + f.z + f.w;
    }
    red[t] = s;
    __syncthreads();
    for (int off = 128; off > 0; off >>= 1) {
        if (t < off) red[t] += red[t + off];
        __syncthreads();
    }
    if (t == 0) dis[row] = rsqrtf(red[0] + 1.0f + 1e-8f);
}

// 2) yT[n][k] = bf16( dis[k] * x[k][n] )   (B-fragment friendly layout)
__global__ void k_prep_y(const float* __restrict__ x, const float* __restrict__ dis,
                         unsigned short* __restrict__ yT) {
    const int g = blockIdx.x * 256 + threadIdx.x;
    const int k = g >> 7, n = g & 127;
    yT[(size_t)n * NROWS + k] = f32_to_bf16(dis[k] * x[g]);
}

// 3) wT[n][k] = bf16( W[k][n] )
__global__ void k_prep_w(const float* __restrict__ w, unsigned short* __restrict__ wT) {
    const int g = blockIdx.x * 256 + threadIdx.x;
    const int k = g >> 7, n = g & 127;
    wT[n * NF + k] = f32_to_bf16(w[g]);
}

// ---------------------------------------------------------------------------
// 4) sbf[i][:] = bf16( dis[i] * ( (adj @ y)[i,:] + y[i,:] ) )
//    64x128 tile/WG, 8 waves, 4 accums/wave; K-loop of 256 x K_TILE=32.
//    Double-buffered LDS, register-staged adj prefetch, async y copies.
// ---------------------------------------------------------------------------
__global__ void __launch_bounds__(256)
k_spmm(const float* __restrict__ adj,
       const unsigned short* __restrict__ yT,
       const float* __restrict__ dis,
       unsigned short* __restrict__ sbf) {
    __shared__ __align__(32) unsigned short abf[2][64 * 32];     // adj tile bf16
    __shared__ __align__(32) unsigned short ytile[2][128 * 32];  // yT tile (n-major)

    const int t    = threadIdx.x;
    const int row0 = blockIdx.x * 64;
    const int lane = t & 31;
    const int wave = t >> 5;
    const int wm   = wave & 3;        // row group of 16
    const int wn   = wave >> 2;       // col half of 64
    const int lm   = lane & 15;
    const int lhi  = lane >> 4;

    const int am = t >> 2,  acb = (t & 3) * 8;    // adj fill: 8 floats/thread
    const int yn = t >> 1,  ykb = (t & 1) * 16;   // y fill : 16 bf16/thread

    const float*          arow = adj + (size_t)(row0 + am) * NROWS + acb;
    const unsigned short* yrow = yT + (size_t)yn * NROWS + ykb;

    v8f acc[4] = {};

    // prologue: stage tile 0 into buffer 0
    {
        v4f g0 = __builtin_nontemporal_load((const v4f*)arow);
        v4f g1 = __builtin_nontemporal_load(((const v4f*)arow) + 1);
        fill_y32B(&ytile[0][yn * 32 + ykb], yrow);
        uint4 pk;
        pk.x = pack_bf16x2(g0.x, g0.y); pk.y = pack_bf16x2(g0.z, g0.w);
        pk.z = pack_bf16x2(g1.x, g1.y); pk.w = pack_bf16x2(g1.z, g1.w);
        *(uint4*)&abf[0][am * 32 + acb] = pk;
    }
    wait_async_and_barrier();

    const int NSTEP = NROWS / 32;
    for (int kt = 0; kt < NSTEP; ++kt) {
        const int  p    = kt & 1;
        const bool more = (kt + 1) < NSTEP;

        // issue next tile's loads before computing on the current one
        v4f g0 = {}, g1 = {};
        if (more) {
            const float* ap = arow + (size_t)(kt + 1) * 32;
            g0 = __builtin_nontemporal_load((const v4f*)ap);
            g1 = __builtin_nontemporal_load(((const v4f*)ap) + 1);
            fill_y32B(&ytile[p ^ 1][yn * 32 + ykb], yrow + (size_t)(kt + 1) * 32);
            if (kt + 2 < NSTEP) __builtin_prefetch(ap + 32, 0, 1);
        }

        // compute on buffer p
        {
            const int kb = lhi * 16;
            union { struct { uint4 lo; uint4 hi; } q; v16bf v; } af;
            af.q.lo = *(const uint4*)&abf[p][(wm * 16 + lm) * 32 + kb];
            af.q.hi = *(const uint4*)&abf[p][(wm * 16 + lm) * 32 + kb + 8];
#pragma unroll
            for (int nt = 0; nt < 4; ++nt) {
                const int n = wn * 64 + nt * 16 + lm;
                v16bf bf = *(const v16bf*)&ytile[p][n * 32 + lhi * 16];
                acc[nt] = __builtin_amdgcn_wmma_f32_16x16x32_bf16(
                    false, af.v, false, bf, (short)0, acc[nt], false, false);
            }
        }

        // convert + store the staged adj tile into the other buffer
        if (more) {
            uint4 pk;
            pk.x = pack_bf16x2(g0.x, g0.y); pk.y = pack_bf16x2(g0.z, g0.w);
            pk.z = pack_bf16x2(g1.x, g1.y); pk.w = pack_bf16x2(g1.z, g1.w);
            *(uint4*)&abf[p ^ 1][am * 32 + acb] = pk;
        }
        wait_async_and_barrier();
    }

    // epilogue: identity add + dis row-scale, store bf16 (L2-resident)
    const int rbase = row0 + wm * 16 + lhi * 8;      // C layout: M = r + 8*lhi
#pragma unroll
    for (int nt = 0; nt < 4; ++nt) {
        const int col = wn * 64 + nt * 16 + lm;
#pragma unroll
        for (int r = 0; r < 8; ++r) {
            const int row = rbase + r;
            const float yv = bf16_to_f32(yT[(size_t)col * NROWS + row]);
            const float sv = dis[row] * (acc[nt][r] + yv);
            sbf[(size_t)row * NF + col] = f32_to_bf16(sv);
        }
    }
}

// ---------------------------------------------------------------------------
// 5) out = relu( s @ W )   (8192x128 @ 128x128, bf16 WMMA, K=128 in 4 steps)
// ---------------------------------------------------------------------------
__global__ void __launch_bounds__(256)
k_out(const unsigned short* __restrict__ sbf,
      const unsigned short* __restrict__ wT,
      float* __restrict__ out) {
    __shared__ __align__(32) unsigned short atile[64 * 128];
    __shared__ __align__(32) unsigned short wtile[128 * 128];

    const int t    = threadIdx.x;
    const int row0 = blockIdx.x * 64;
    {
        const int m = t >> 2, cb = (t & 3) * 32;
        const uint4* sp = (const uint4*)(sbf + (size_t)(row0 + m) * NF + cb);
        uint4* dp = (uint4*)&atile[m * 128 + cb];
#pragma unroll
        for (int j = 0; j < 4; ++j) dp[j] = sp[j];
    }
    {
        const int n = t >> 1, kb = (t & 1) * 64;
        const uint4* wp = (const uint4*)(wT + n * NF + kb);
        uint4* dp = (uint4*)&wtile[n * 128 + kb];
#pragma unroll
        for (int j = 0; j < 8; ++j) dp[j] = wp[j];
    }
    __syncthreads();

    const int lane = t & 31, wave = t >> 5;
    const int wm = wave & 3, wn = wave >> 2;
    const int lm = lane & 15, lhi = lane >> 4;

    v8f acc[4] = {};
#pragma unroll
    for (int kt = 0; kt < 4; ++kt) {
        const int k0 = kt * 32;
        union { struct { uint4 lo; uint4 hi; } q; v16bf v; } af;
        const int kb = k0 + lhi * 16;
        af.q.lo = *(const uint4*)&atile[(wm * 16 + lm) * 128 + kb];
        af.q.hi = *(const uint4*)&atile[(wm * 16 + lm) * 128 + kb + 8];
#pragma unroll
        for (int nt = 0; nt < 4; ++nt) {
            const int n = wn * 64 + nt * 16 + lm;
            v16bf bf = *(const v16bf*)&wtile[n * 128 + k0 + lhi * 16];
            acc[nt] = __builtin_amdgcn_wmma_f32_16x16x32_bf16(
                false, af.v, false, bf, (short)0, acc[nt], false, false);
        }
    }

    const int rbase = row0 + wm * 16 + lhi * 8;
#pragma unroll
    for (int nt = 0; nt < 4; ++nt) {
        const int col = wn * 64 + nt * 16 + lm;
#pragma unroll
        for (int r = 0; r < 8; ++r)
            __builtin_nontemporal_store(fmaxf(acc[nt][r], 0.0f),
                                        &out[(size_t)(rbase + r) * NF + col]);
    }
}

// ---------------------------------------------------------------------------
extern "C" void kernel_launch(void* const* d_in, const int* in_sizes, int n_in,
                              void* d_out, int out_size, void* d_ws, size_t ws_size,
                              hipStream_t stream) {
    (void)in_sizes; (void)n_in; (void)out_size; (void)ws_size;
    const float* x   = (const float*)d_in[0];   // [8192,128]
    const float* adj = (const float*)d_in[1];   // [8192,8192]
    const float* w   = (const float*)d_in[2];   // [128,128]
    float* out = (float*)d_out;                 // [8192,128]

    unsigned char* ws = (unsigned char*)d_ws;
    float*          dis = (float*)ws;                                          // 32 KB
    unsigned short* yT  = (unsigned short*)(ws + (32u << 10));                 // 2 MB
    unsigned short* wT  = (unsigned short*)(ws + (32u << 10) + (2u << 20));    // 32 KB
    unsigned short* sbf = (unsigned short*)(ws + (64u << 10) + (2u << 20));    // 2 MB

    k_deg   <<<NROWS,            256, 0, stream>>>(adj, dis);
    k_prep_y<<<(NROWS * NF)/256, 256, 0, stream>>>(x, dis, yT);
    k_prep_w<<<(NF * NF)/256,    256, 0, stream>>>(w, wT);
    k_spmm  <<<NROWS / 64,       256, 0, stream>>>(adj, yT, dis, sbf);
    k_out   <<<NROWS / 64,       256, 0, stream>>>(sbf, wT, out);
}
